// Attention_Layer_69037304316023
// MI455X (gfx1250) — matrix operand order repaired and checked by reading, not compiled, not run on hardware
//
#include <hip/hip_runtime.h>
#include <hip/hip_bf16.h>

typedef __attribute__((ext_vector_type(16))) _Float16 v16h;
typedef __attribute__((ext_vector_type(8)))  _Float16 half8;
typedef __attribute__((ext_vector_type(8)))  float    v8f;
typedef __attribute__((ext_vector_type(4)))  float    f32x4;
typedef int i32x4 __attribute__((vector_size(16)));
typedef __attribute__((address_space(1))) i32x4* gas_i32x4_p;
typedef __attribute__((address_space(3))) i32x4* las_i32x4_p;

#define BB     32
#define SS     2048
#define DK     1024   // D_ENC = K
#define NU     1024   // UNITS = N
#define MT     64     // rows per workgroup s-tile
#define NT     32     // cols per iteration (double-buffered)
#define SA     1032   // padded LDS stride (halves): 16B-aligned rows + conflict-free banks
#define SB     1032
#define NITER  (NU / NT)

#if defined(__has_builtin)
#  if __has_builtin(__builtin_amdgcn_global_load_async_to_lds_b128)
#    define HAVE_ASYNC_LDS 1
#  endif
#  if __has_builtin(__builtin_amdgcn_s_wait_asynccnt)
#    define HAVE_WAIT_ASYNC 1
#  endif
#endif
#ifndef HAVE_ASYNC_LDS
#  define HAVE_ASYNC_LDS 0
#endif
#ifndef HAVE_WAIT_ASYNC
#  define HAVE_WAIT_ASYNC 0
#endif

// ---------------------------------------------------------------------------
// 1) W1_enc (f32, [k][u]) -> W1T (f16, [u][k]) via LDS tile transpose
// ---------------------------------------------------------------------------
__global__ __launch_bounds__(256) void w1_transpose_kernel(const float* __restrict__ W1,
                                                           _Float16* __restrict__ w1t) {
    __shared__ _Float16 tile[64 * 65];
    const int bx  = blockIdx.x;
    const int k0  = (bx & 15) * 64;
    const int u0  = (bx >> 4) * 64;
    const int t   = threadIdx.x;
    const int col = t & 63;
    const int rr  = t >> 6;
#pragma unroll
    for (int i = 0; i < 16; ++i) {
        int row = i * 4 + rr;   // k_local
        tile[col * 65 + row] = (_Float16)W1[(size_t)(k0 + row) * NU + u0 + col];
    }
    __syncthreads();
#pragma unroll
    for (int i = 0; i < 16; ++i) {
        int row = i * 4 + rr;   // u_local
        w1t[(size_t)(u0 + row) * DK + k0 + col] = tile[row * 65 + col];
    }
}

// ---------------------------------------------------------------------------
// 2) dec_proj[b][u] = h_dec[b] . W1_dec[:,u] + b1[u]
// ---------------------------------------------------------------------------
__global__ __launch_bounds__(256) void dec_proj_kernel(const float* __restrict__ h_dec,
                                                       const float* __restrict__ W1,
                                                       const float* __restrict__ b1,
                                                       float* __restrict__ dp) {
    const int t = blockIdx.x * 256 + threadIdx.x;
    const int b = t >> 10;
    const int u = t & 1023;
    const float* wcol = W1 + (size_t)DK * NU + u;
    const float* hv   = h_dec + b * DK;
    float s = b1[u];
    for (int k = 0; k < DK; ++k)
        s = fmaf(hv[k], wcol[(size_t)k * NU], s);
    dp[t] = s;
}

__device__ __forceinline__ float fast_tanh(float x) {
    float e = __expf(2.0f * x);
    return 1.0f - 2.0f * __builtin_amdgcn_rcpf(e + 1.0f);
}

__device__ __forceinline__ v16h lda_frag(const _Float16* p, int k0) {
    half8 x0 = *(const half8*)(p + k0);        // K 0..7   (of this lane's half)
    half8 x1 = *(const half8*)(p + k0 + 16);   // K 16..23
    return __builtin_shufflevector(x0, x1, 0,1,2,3,4,5,6,7,8,9,10,11,12,13,14,15);
}
__device__ __forceinline__ v16h ldb_frag(const _Float16* p, int k0) {
    half8 x0 = *(const half8*)(p + k0);
    half8 x1 = *(const half8*)(p + k0 + 16);
    return __builtin_shufflevector(x0, x1, 0,1,2,3,4,5,6,7,8,9,10,11,12,13,14,15);
}

// cooperative B stage: NT rows of W1T -> LDS (async direct-to-LDS when available)
__device__ __forceinline__ void stage_B(_Float16* dst, const _Float16* __restrict__ w1t,
                                        int nbase, int tid) {
    for (int i = tid; i < NT * (DK / 8); i += 128) {
        int row = i >> 7;            // DK/8 = 128 chunks per row
        int c8  = i & 127;
        const _Float16* g = w1t + (size_t)(nbase + row) * DK + c8 * 8;
        _Float16* l = dst + row * SB + c8 * 8;
#if HAVE_ASYNC_LDS
        __builtin_amdgcn_global_load_async_to_lds_b128(
            (gas_i32x4_p)(i32x4*)(void*)const_cast<_Float16*>(g),
            (las_i32x4_p)(i32x4*)(void*)l, 0, 0);
#else
        *(half8*)l = *(const half8*)g;
#endif
    }
}

__device__ __forceinline__ void wait_async_lds() {
#if HAVE_ASYNC_LDS
#  if HAVE_WAIT_ASYNC
    __builtin_amdgcn_s_wait_asynccnt(0);
#  else
    asm volatile("s_wait_asynccnt 0" ::: "memory");
#  endif
#endif
}

// ---------------------------------------------------------------------------
// 3) scores[b][s] = relu( tanh(x @ W1enc + dec_proj) @ W2 + b2 )
//    4 waves (2M x 2N); each wave: 32x16 tile, pipelined frags, dbl-buffered B.
// ---------------------------------------------------------------------------
__global__ __launch_bounds__(128, 1) void attn_scores_kernel(const float* __restrict__ h_enc,
                                                             const _Float16* __restrict__ w1t,
                                                             const float* __restrict__ dec_proj,
                                                             const float* __restrict__ w2,
                                                             const float* __restrict__ b2,
                                                             float* __restrict__ scores) {
    extern __shared__ __align__(16) _Float16 smem[];
    _Float16* Alds = smem;                       // MT x SA
    _Float16* Bl0  = smem + MT * SA;             // NT x SB (buffer 0)
    _Float16* Bl1  = Bl0 + NT * SB;              // NT x SB (buffer 1)
    float*    sred = (float*)(smem + MT * SA + 2 * NT * SB);   // [2][MT]

    const int tid  = threadIdx.x;
    const int lane = tid & 31;
    const int wave = tid >> 5;
    const int mw   = wave & 1;                   // M half (rows)
    const int nw   = wave >> 1;                  // N half (cols)
    const int b    = blockIdx.x >> 5;            // 32 s-tiles per batch
    const int s0   = (blockIdx.x & 31) * MT;

    // --- stage A: MT x DK f32 -> f16 LDS ---
    const float* Ag = h_enc + ((size_t)b * SS + s0) * DK;
    for (int i = tid; i < MT * (DK / 8); i += 128) {
        int row = i >> 7;
        int c8  = i & 127;
        const float* g = Ag + (size_t)row * DK + c8 * 8;
        f32x4 v0 = *(const f32x4*)g;
        f32x4 v1 = *(const f32x4*)(g + 4);
        half8 h;
        h[0]=(_Float16)v0.x; h[1]=(_Float16)v0.y; h[2]=(_Float16)v0.z; h[3]=(_Float16)v0.w;
        h[4]=(_Float16)v1.x; h[5]=(_Float16)v1.y; h[6]=(_Float16)v1.z; h[7]=(_Float16)v1.w;
        *(half8*)(Alds + row * SA + c8 * 8) = h;
    }
    stage_B(Bl0, w1t, 0, tid);                   // prefetch first B block
    wait_async_lds();
    __syncthreads();

    const int m    = lane & 15;
    const int ksel = lane >> 4;
    const int Rw   = mw * 32;
    const _Float16* arow0 = Alds + (Rw + m) * SA + ksel * 8;
    const _Float16* arow1 = Alds + (Rw + 16 + m) * SA + ksel * 8;

    float partial[16];
#pragma unroll
    for (int i = 0; i < 16; ++i) partial[i] = 0.0f;

    for (int nb = 0; nb < NITER; ++nb) {
        _Float16* Bcur  = (nb & 1) ? Bl1 : Bl0;
        _Float16* Bnext = (nb & 1) ? Bl0 : Bl1;
        if (nb + 1 < NITER) stage_B(Bnext, w1t, (nb + 1) * NT, tid);  // overlaps compute

        const _Float16* brow = Bcur + (nw * 16 + m) * SB + ksel * 8;

        v8f c0 = {0.f,0.f,0.f,0.f,0.f,0.f,0.f,0.f};
        v8f c1 = {0.f,0.f,0.f,0.f,0.f,0.f,0.f,0.f};
        // software-pipelined K loop: load k+1 frags before issuing k's WMMAs
        v16h a0 = lda_frag(arow0, 0);
        v16h a1 = lda_frag(arow1, 0);
        v16h bf = ldb_frag(brow, 0);
        for (int k0 = 32; k0 < DK; k0 += 32) {
            v16h na0 = lda_frag(arow0, k0);
            v16h na1 = lda_frag(arow1, k0);
            v16h nbf = ldb_frag(brow, k0);
            c0 = __builtin_amdgcn_wmma_f32_16x16x32_f16(false, a0, false, bf, (short)0, c0, false, false);
            c1 = __builtin_amdgcn_wmma_f32_16x16x32_f16(false, a1, false, bf, (short)0, c1, false, false);
            a0 = na0; a1 = na1; bf = nbf;
        }
        c0 = __builtin_amdgcn_wmma_f32_16x16x32_f16(false, a0, false, bf, (short)0, c0, false, false);
        c1 = __builtin_amdgcn_wmma_f32_16x16x32_f16(false, a1, false, bf, (short)0, c1, false, false);

        // fused epilogue: tanh + W2 partial dot (this lane's column n)
        const int n  = nb * NT + nw * 16 + m;
        float dp  = dec_proj[b * NU + n];        // includes b1
        float w2n = w2[n];
#pragma unroll
        for (int r = 0; r < 8; ++r) {
            partial[r]     = fmaf(fast_tanh(c0[r] + dp), w2n, partial[r]);     // row Rw+r+8ksel
            partial[8 + r] = fmaf(fast_tanh(c1[r] + dp), w2n, partial[8 + r]); // row Rw+16+r+8ksel
        }

        wait_async_lds();     // own prefetch landed
        __syncthreads();      // all waves done with Bcur; Bnext valid for everyone
    }

    // reduce 16 N-lanes of each half-wave, then combine the two N-wave halves via LDS
#pragma unroll
    for (int idx = 0; idx < 16; ++idx) {
        float p = partial[idx];
        for (int off = 1; off < 16; off <<= 1)
            p += __shfl_xor(p, off, 16);
        if (m == 0) {
            int row = Rw + (idx >> 3) * 16 + (idx & 7) + 8 * ksel;
            sred[nw * MT + row] = p;
        }
    }
    __syncthreads();
    if (tid < MT) {
        float v = sred[tid] + sred[MT + tid] + b2[0];
        scores[(size_t)b * SS + s0 + tid] = fmaxf(v, 0.0f);   // relu
    }
}

// ---------------------------------------------------------------------------
// 4) softmax over S + context vector; one workgroup per batch
//    out: [0, 32768) context (B x 1024); [32768, 98304) weights (B x 2048)
// ---------------------------------------------------------------------------
__global__ __launch_bounds__(256) void attn_softmax_ctx_kernel(const float* __restrict__ h_enc,
                                                               const float* __restrict__ scores,
                                                               float* __restrict__ out) {
    __shared__ float w_lds[SS];
    __shared__ float red[256];
    const int b   = blockIdx.x;
    const int tid = threadIdx.x;
    const float* sc = scores + (size_t)b * SS;

    float v[8];
    float mx = -1e30f;
#pragma unroll
    for (int i = 0; i < 8; ++i) { v[i] = sc[tid + 256 * i]; mx = fmaxf(mx, v[i]); }
    red[tid] = mx; __syncthreads();
    for (int off = 128; off >= 1; off >>= 1) {
        if (tid < off) red[tid] = fmaxf(red[tid], red[tid + off]);
        __syncthreads();
    }
    mx = red[0]; __syncthreads();

    float sum = 0.0f;
#pragma unroll
    for (int i = 0; i < 8; ++i) { v[i] = __expf(v[i] - mx); sum += v[i]; }
    red[tid] = sum; __syncthreads();
    for (int off = 128; off >= 1; off >>= 1) {
        if (tid < off) red[tid] += red[tid + off];
        __syncthreads();
    }
    float inv = 1.0f / red[0];
    __syncthreads();

#pragma unroll
    for (int i = 0; i < 8; ++i) {
        float w = v[i] * inv;
        w_lds[tid + 256 * i] = w;
        out[BB * DK + (size_t)b * SS + tid + 256 * i] = w;
    }
    __syncthreads();

    f32x4 acc = {0.f, 0.f, 0.f, 0.f};
    const float* hb = h_enc + (size_t)b * SS * DK;
    for (int s = 0; s < SS; ++s) {
        float w = w_lds[s];
        f32x4 x = *(const f32x4*)(hb + (size_t)s * DK + tid * 4);
        acc += x * w;
    }
    *(f32x4*)(out + (size_t)b * DK + tid * 4) = acc;
}

// ---------------------------------------------------------------------------
extern "C" void kernel_launch(void* const* d_in, const int* in_sizes, int n_in,
                              void* d_out, int out_size, void* d_ws, size_t ws_size,
                              hipStream_t stream) {
    (void)in_sizes; (void)n_in; (void)out_size; (void)ws_size;
    const float* h_enc = (const float*)d_in[0];
    const float* h_dec = (const float*)d_in[1];
    const float* W1    = (const float*)d_in[2];
    const float* b1    = (const float*)d_in[3];
    const float* W2    = (const float*)d_in[4];
    const float* b2    = (const float*)d_in[5];
    float* out = (float*)d_out;

    char* ws = (char*)d_ws;
    _Float16* w1t  = (_Float16*)ws;                                          // 2 MB
    float* dp      = (float*)(ws + (size_t)DK * NU * sizeof(_Float16));      // 128 KB
    float* scoresb = (float*)(ws + (size_t)DK * NU * sizeof(_Float16)
                                 + (size_t)BB * NU * sizeof(float));         // 256 KB

    w1_transpose_kernel<<<256, 256, 0, stream>>>(W1, w1t);
    dec_proj_kernel<<<BB * NU / 256, 256, 0, stream>>>(h_dec, W1, b1, dp);

    size_t shmem = (size_t)(MT * SA + 2 * NT * SB) * sizeof(_Float16)
                 + 2 * MT * sizeof(float);                                   // ~259 KB
    attn_scores_kernel<<<BB * (SS / MT), 128, shmem, stream>>>(h_enc, w1t, dp, W2, b2, scoresb);

    attn_softmax_ctx_kernel<<<BB, 256, 0, stream>>>(h_enc, scoresb, out);
}
